// CrossAttentionBottleneck_7413113553383
// MI455X (gfx1250) — matrix operand order repaired and checked by reading, not compiled, not run on hardware
//
#include <hip/hip_runtime.h>

// ---------------------------------------------------------------------------
// Types
// ---------------------------------------------------------------------------
typedef _Float16 half_t;
typedef _Float16 v16h __attribute__((ext_vector_type(16)));
typedef _Float16 v8h  __attribute__((ext_vector_type(8)));
typedef _Float16 v4h  __attribute__((ext_vector_type(4)));
typedef float    v8f  __attribute__((ext_vector_type(8)));
typedef float    v4f  __attribute__((ext_vector_type(4)));
typedef unsigned int       u32;
typedef unsigned long long u64;
typedef u32 v4u __attribute__((ext_vector_type(4)));
typedef int v8i __attribute__((ext_vector_type(8)));
typedef int v4i __attribute__((ext_vector_type(4)));

#ifndef __has_builtin
#define __has_builtin(x) 0
#endif
#if __has_builtin(__builtin_amdgcn_tensor_load_to_lds) && __has_builtin(__builtin_amdgcn_s_wait_tensorcnt)
#define USE_TDM 1
#else
#define USE_TDM 0
#endif

#define TDM_PUBLISH_FENCE() asm volatile("" ::: "memory")

// The TDM engine writes LDS behind the compiler's back (the builtin carries no
// LDS pointer), so module-wide the staging buffer looks never-stored and LLVM
// folds every fragment load to undef.  This helper plants an opaque,
// dynamically-dead may-store (device VAs never have bits[63:61]==0b101) with
// a loop-variant opaque index: it defeats both undef-folding and
// cross-iteration CSE of the double-buffered LDS reads, at zero runtime cost.
__device__ __forceinline__ void tdm_guard(char* s, u32 bytes, int iter, const void* key)
{
    const u64 kv = (u64)key;
    if ((kv >> 61) == 5ull)
        s[((u32)kv + (u32)iter * 1237u) % bytes] = (char)kv;
}

// Problem constants
#define BATCH   32
#define CH      512
#define NPIX    1024      // 32*32
#define HEADS   8
#define HEAD_CH 64

// ---------------------------------------------------------------------------
// TDM: 2D tile load, tile width fixed at 32 f16 elements (= 16 dwords/row).
// pad_enable with pad_interval=3 (16 dwords) and pad_amount=7 (8 dwords)
// gives an LDS row pitch of 48 halves (96 B) -> conflict-friendly, 16B-aligned
// fragment reads.  Descriptor packing per CDNA5 ISA.
// 6-operand builtin form: (uint32x4 g0, int32x8 g1, int32x4 g2, int32x4 g3,
//                          int32x8 extra, i32 cpol); extra is zeroed.
// ---------------------------------------------------------------------------
#if USE_TDM
__device__ __forceinline__ void tdm_tile32_f16(u64 gaddr, u32 lds_off, u32 rows,
                                               u64 row_stride_elems,
                                               u32 tdim0, u32 tdim1)
{
    v4u g0;
    g0.x = 1u;                                   // count=1, user mode, no gather
    g0.y = lds_off;                              // LDS byte address
    g0.z = (u32)gaddr;                           // global_addr[31:0]
    g0.w = ((u32)(gaddr >> 32) & 0x01FFFFFFu) | (2u << 30);  // addr[56:32], type=2
    const u32 w0 = (1u << 16)    // data_size = 1 (2 bytes)
                 | (1u << 20)    // pad_enable
                 | (3u << 22)    // pad_interval: 16 dwords (one 32-half row)
                 | (7u << 25);   // pad_amount : 8 dwords  -> 96B pitch
    const u32 w1 = (tdim0 & 0xFFFFu) << 16;
    const u32 w2 = (tdim0 >> 16) | ((tdim1 & 0xFFFFu) << 16);
    const u32 w3 = (tdim1 >> 16) | (32u << 16);             // tile_dim0 = 32
    const u32 w4 = rows & 0xFFFFu;                          // tile_dim1 ; tile_dim2=0
    const u32 w5 = (u32)row_stride_elems;                   // dim0 stride [31:0]
    const u32 w6 = (u32)(row_stride_elems >> 32) & 0xFFFFu; // dim0 stride [47:32]
    v8i g1 = { (int)w0, (int)w1, (int)w2, (int)w3, (int)w4, (int)w5, (int)w6, 0 };
    v4i g2 = { 1, 1, 0, 0 };   // tensor_dim2=1, tensor_dim3=1 (2D tile)
    v4i g3 = { 0, 0, 0, 0 };
    v8i g4 = { 0, 0, 0, 0, 0, 0, 0, 0 };   // beyond documented D#: zero
    __builtin_amdgcn_tensor_load_to_lds(g0, g1, g2, g3, g4, 0);
}
#endif

// Fallback cooperative stage of a rows x 32-half tile into 48-half-pitch LDS.
__device__ __forceinline__ void stage_tile32(const half_t* __restrict__ g, half_t* s,
                                             int rows, long strideElems, int tid)
{
    for (int u = tid; u < rows * 2; u += 256) {
        const int row = u >> 1, ch = (u & 1) * 16;
        const v8h* src = (const v8h*)(g + (long)row * strideElems + ch);
        v8h a = src[0], b = src[1];
        *(v8h*)(s + row * 48 + ch)     = a;
        *(v8h*)(s + row * 48 + ch + 8) = b;
    }
}

// ---------------------------------------------------------------------------
// WMMA helpers (layouts per CDNA5 ISA §7.12.2, wave32)
//   A (16x32 f16): lane row m = lane&15, K runs [g*8,+8) and [16+g*8,+8)
//   B (32x16 f16): lane col n = lane&15, K runs [g*8,+8) and [16+g*8,+8)
//   D (16x16 f32): lane n = lane&15 fixed, M = (lane>>4)*8 + r (r contiguous)
// ---------------------------------------------------------------------------
__device__ __forceinline__ v8f wmma_f16(v16h a, v16h b, v8f c)
{
    return __builtin_amdgcn_wmma_f32_16x16x32_f16(false, a, false, b,
                                                  (short)0, c, false, false);
}
__device__ __forceinline__ v16h load_a_frag(const half_t* s, int row, int g)
{
    const v8h lo = *(const v8h*)(s + row * 48 + g * 8);
    const v8h hi = *(const v8h*)(s + row * 48 + 16 + g * 8);
    return __builtin_shufflevector(lo, hi, 0,1,2,3,4,5,6,7,8,9,10,11,12,13,14,15);
}
__device__ __forceinline__ v16h load_b_frag(const half_t* s, int row, int g)
{
    const v8h lo = *(const v8h*)(s + row * 48 + g * 8);
    const v8h hi = *(const v8h*)(s + row * 48 + 16 + g * 8);
    return __builtin_shufflevector(lo, hi, 0,1,2,3,4,5,6,7,8,9,10,11,12,13,14,15);
}

__device__ __forceinline__ float elu_feat(float x)
{   // clip(elu(x)+1, -10, 10): x>0 -> min(x+1,10); else exp(x) in (0,1]
    return x > 0.f ? fminf(x + 1.f, 10.f) : __expf(x);
}

// ---------------------------------------------------------------------------
// Pre-pass: f32 [b][c][p] -> f16 [b][p][c] with clip(+-20), via LDS transpose
// ---------------------------------------------------------------------------
__global__ __launch_bounds__(256)
void txc_kernel(const float* __restrict__ x, half_t* __restrict__ xt)
{
    __shared__ half_t tile[32 * 33];
    const int tid = threadIdx.x;
    const int b = blockIdx.z, cB = blockIdx.y * 32, pB = blockIdx.x * 32;
    {
        const int c = tid >> 3, p4 = (tid & 7) * 4;
        v4f v = *(const v4f*)(x + ((size_t)(b * CH + cB + c)) * NPIX + pB + p4);
        for (int j = 0; j < 4; ++j) {
            float f = fminf(fmaxf(v[j], -20.f), 20.f);
            tile[(p4 + j) * 33 + c] = (half_t)f;
        }
    }
    __syncthreads();
    {
        const int p = tid >> 3, c4 = (tid & 7) * 4;
        v4h o;
        for (int j = 0; j < 4; ++j) o[j] = tile[p * 33 + c4 + j];
        *(v4h*)(xt + ((size_t)(b * NPIX + pB + p)) * CH + cB + c4) = o;
    }
}

// f32 -> f16 elementwise (weights)
__global__ __launch_bounds__(256)
void cvt_kernel(const float* __restrict__ src, half_t* __restrict__ dst, int n4)
{
    const int i = blockIdx.x * 256 + threadIdx.x;
    if (i < n4) {
        v4f v = *(const v4f*)(src + (size_t)i * 4);
        v4h o;
        for (int j = 0; j < 4; ++j) o[j] = (half_t)v[j];
        *(v4h*)(dst + (size_t)i * 4) = o;
    }
}

__global__ __launch_bounds__(256)
void zero_kernel(float* __restrict__ p, int n)
{
    const int i = blockIdx.x * 256 + threadIdx.x;
    if (i < n) p[i] = 0.f;
}

// ---------------------------------------------------------------------------
// GEMM: out[o,p] = sum_c W16[o,c] * Xin[b,p,c]  (+bias, mode epilogue)
//   MODE 0 (Q): elu_feat(v*0.125) -> qf [b][h][p][64]   (b128 transposed store)
//   MODE 1 (K): elu_feat(v)       -> kf [b][c][p]       (scatter f16)
//   MODE 2 (V): v                 -> vv [b][c][p]       (scatter f16)
//   MODE 3 (O): v -> proj f32 [b][c][p] + group sum/sumsq atomics
// Block = 256 thr (8 waves), tile 128(o) x 128(p), K-step 32, double-buffered
// TDM feeds both A (weights) and B (activations) tiles.
// ---------------------------------------------------------------------------
template<int MODE>
__global__ __launch_bounds__(256)
void gemm512_kernel(const half_t* __restrict__ W16, const half_t* __restrict__ Xin,
                    const float* __restrict__ bias, void* __restrict__ outp,
                    float* __restrict__ stats)
{
    __shared__ __align__(128) char smem[49152];   // 2 x (A 12288 + B 12288)
    const int tid = threadIdx.x;
    const int lane = tid & 31, wave = tid >> 5;
    const int wm = wave >> 2, wn = wave & 3;
    const int g = lane >> 4, ln = lane & 15;
    const int b = blockIdx.z;
    const int oBase = blockIdx.y * 128;
    const int pBase = blockIdx.x * 128;

    const half_t* gW = W16 + (size_t)oBase * CH;
    const half_t* gX = Xin + ((size_t)(b * NPIX + pBase)) * CH;

    v8f acc[4][2];
    const v8f z8 = {0.f,0.f,0.f,0.f,0.f,0.f,0.f,0.f};
    for (int i = 0; i < 4; ++i) for (int j = 0; j < 2; ++j) acc[i][j] = z8;

#if USE_TDM
    if (wave == 0) {
        tdm_tile32_f16((u64)gW, 0u,      128, CH, CH, CH);
        tdm_tile32_f16((u64)gX, 12288u,  128, CH, CH, NPIX);
    }
#else
    stage_tile32(gW, (half_t*)(smem + 0),     128, CH, tid);
    stage_tile32(gX, (half_t*)(smem + 12288), 128, CH, tid);
#endif

    for (int k = 0; k < 16; ++k) {
#if USE_TDM
        if (wave == 0) {
            if (k + 1 < 16) {
                const u32 lo = (k & 1) ? 0u : 24576u;
                tdm_tile32_f16((u64)(gW + (k + 1) * 32), lo,          128, CH, CH, CH);
                tdm_tile32_f16((u64)(gX + (k + 1) * 32), lo + 12288u, 128, CH, CH, NPIX);
                __builtin_amdgcn_s_wait_tensorcnt(2);   // pair k complete
            } else {
                __builtin_amdgcn_s_wait_tensorcnt(0);
            }
        }
#endif
#if !USE_TDM
        if (k + 1 < 16) {
            char* nb = smem + (((k + 1) & 1) ? 24576 : 0);
            stage_tile32(gW + (k + 1) * 32, (half_t*)nb,           128, CH, tid);
            stage_tile32(gX + (k + 1) * 32, (half_t*)(nb + 12288), 128, CH, tid);
        }
#endif
        __syncthreads();
#if USE_TDM
        tdm_guard(smem, 49152u, k, W16);   // opaque dead store: TDM wrote smem
        TDM_PUBLISH_FENCE();
#endif
        const half_t* sA = (const half_t*)(smem + ((k & 1) ? 24576 : 0));
        const half_t* sB = (const half_t*)((const char*)sA + 12288);

        const v16h bf0 = load_b_frag(sB, wn * 32 + ln, g);
        const v16h bf1 = load_b_frag(sB, wn * 32 + 16 + ln, g);
#pragma unroll
        for (int mt = 0; mt < 4; ++mt) {
            const v16h af = load_a_frag(sA, wm * 64 + mt * 16 + ln, g);
            acc[mt][0] = wmma_f16(af, bf0, acc[mt][0]);
            acc[mt][1] = wmma_f16(af, bf1, acc[mt][1]);
        }
        __syncthreads();
    }

    // ---- epilogue ----
#pragma unroll
    for (int mt = 0; mt < 4; ++mt) {
        const int o0 = oBase + wm * 64 + mt * 16 + g * 8;   // 8 contiguous out chans
        float bv[8];
        {
            v4f b0 = *(const v4f*)(bias + o0);
            v4f b1 = *(const v4f*)(bias + o0 + 4);
            bv[0]=b0.x; bv[1]=b0.y; bv[2]=b0.z; bv[3]=b0.w;
            bv[4]=b1.x; bv[5]=b1.y; bv[6]=b1.z; bv[7]=b1.w;
        }
        if constexpr (MODE == 3) {
            float* proj = (float*)outp;
            float s = 0.f, ss = 0.f;
#pragma unroll
            for (int nt = 0; nt < 2; ++nt) {
                const int p = pBase + wn * 32 + nt * 16 + ln;
#pragma unroll
                for (int r = 0; r < 8; ++r) {
                    float v = acc[mt][nt][r] + bv[r];
                    proj[((size_t)(b * CH + o0 + r)) * NPIX + p] = v;
                    s += v; ss += v * v;
                }
            }
            for (int off = 16; off > 0; off >>= 1) {
                s  += __shfl_xor(s,  off, 32);
                ss += __shfl_xor(ss, off, 32);
            }
            if (lane == 0) {   // each m-tile == one 16-channel GN group
                const int grp = (oBase + wm * 64 + mt * 16) >> 4;
                atomicAdd(&stats[(b * 32 + grp) * 2 + 0], s);
                atomicAdd(&stats[(b * 32 + grp) * 2 + 1], ss);
            }
        } else {
#pragma unroll
            for (int nt = 0; nt < 2; ++nt) {
                const int p = pBase + wn * 32 + nt * 16 + ln;
                if constexpr (MODE == 0) {
                    half_t* qf = (half_t*)outp;
                    const int h = o0 >> 6, d0 = o0 & 63;
                    v8h o8;
                    for (int r = 0; r < 8; ++r)
                        o8[r] = (half_t)elu_feat((acc[mt][nt][r] + bv[r]) * 0.125f);
                    *(v8h*)(qf + ((size_t)((b * HEADS + h) * NPIX + p)) * 64 + d0) = o8;
                } else {
                    half_t* dst = (half_t*)outp;
                    for (int r = 0; r < 8; ++r) {
                        float v = acc[mt][nt][r] + bv[r];
                        if constexpr (MODE == 1) v = elu_feat(v);
                        dst[((size_t)(b * CH + o0 + r)) * NPIX + p] = (half_t)v;
                    }
                }
            }
        }
    }
    (void)stats;
}

// ---------------------------------------------------------------------------
// ksum[bh][d] = sum_p kf[bh*64+d][p]
// ---------------------------------------------------------------------------
__global__ __launch_bounds__(256)
void ksum_kernel(const half_t* __restrict__ kf, float* __restrict__ ksum)
{
    const int bh = blockIdx.x, tid = threadIdx.x;
    const int d = tid & 63, q = tid >> 6;
    const half_t* row = kf + ((size_t)(bh * 64 + d)) * NPIX + q * 256;
    float s = 0.f;
    for (int i = 0; i < 32; ++i) {
        v8h x = *(const v8h*)(row + i * 8);
        for (int j = 0; j < 8; ++j) s += (float)x[j];
    }
    __shared__ float red[256];
    red[q * 64 + d] = s;
    __syncthreads();
    if (tid < 64)
        ksum[(size_t)bh * 64 + tid] =
            red[tid] + red[64 + tid] + red[128 + tid] + red[192 + tid];
}

// ---------------------------------------------------------------------------
// ctx[bh][e][d] = sum_p kf[bh,d,p] * vv[bh,e,p]   (K^T V, stored e-major)
// One block per (b,h); 8 waves x 2 WMMA tiles; 32 K-steps of 32 p, TDM-fed.
// ---------------------------------------------------------------------------
__global__ __launch_bounds__(256)
void ctx_kernel(const half_t* __restrict__ kf, const half_t* __restrict__ vv,
                half_t* __restrict__ ctx)
{
    __shared__ __align__(128) char smem[24576];   // 2 x (K 6144 + V 6144)
    const int tid = threadIdx.x;
    const int lane = tid & 31, wave = tid >> 5;
    const int g = lane >> 4, ln = lane & 15;
    const int bh = blockIdx.x;
    const int dt = wave >> 1, etA = (wave & 1) * 2, etB = etA + 1;

    const half_t* gK = kf + ((size_t)bh * 64) * NPIX;   // (b*512+h*64) rows
    const half_t* gV = vv + ((size_t)bh * 64) * NPIX;

    const v8f z8 = {0.f,0.f,0.f,0.f,0.f,0.f,0.f,0.f};
    v8f accA = z8, accB = z8;

#if USE_TDM
    if (wave == 0) {
        tdm_tile32_f16((u64)gK, 0u,     64, NPIX, NPIX, CH);
        tdm_tile32_f16((u64)gV, 6144u,  64, NPIX, NPIX, CH);
    }
#else
    stage_tile32(gK, (half_t*)(smem + 0),    64, NPIX, tid);
    stage_tile32(gV, (half_t*)(smem + 6144), 64, NPIX, tid);
#endif

    for (int kk = 0; kk < 32; ++kk) {
#if USE_TDM
        if (wave == 0) {
            if (kk + 1 < 32) {
                const u32 lo = (kk & 1) ? 0u : 12288u;
                tdm_tile32_f16((u64)(gK + (kk + 1) * 32), lo,         64, NPIX, NPIX, CH);
                tdm_tile32_f16((u64)(gV + (kk + 1) * 32), lo + 6144u, 64, NPIX, NPIX, CH);
                __builtin_amdgcn_s_wait_tensorcnt(2);
            } else {
                __builtin_amdgcn_s_wait_tensorcnt(0);
            }
        }
#endif
#if !USE_TDM
        if (kk + 1 < 32) {
            char* nb = smem + (((kk + 1) & 1) ? 12288 : 0);
            stage_tile32(gK + (kk + 1) * 32, (half_t*)nb,          64, NPIX, tid);
            stage_tile32(gV + (kk + 1) * 32, (half_t*)(nb + 6144), 64, NPIX, tid);
        }
#endif
        __syncthreads();
#if USE_TDM
        tdm_guard(smem, 24576u, kk, kf);   // opaque dead store: TDM wrote smem
        TDM_PUBLISH_FENCE();
#endif
        const half_t* sK = (const half_t*)(smem + ((kk & 1) ? 12288 : 0));
        const half_t* sV = sK + 3072;

        const v16h a  = load_a_frag(sK, dt * 16 + ln, g);
        const v16h b0 = load_b_frag(sV, etA * 16 + ln, g);
        const v16h b1 = load_b_frag(sV, etB * 16 + ln, g);
        accA = wmma_f16(a, b0, accA);
        accB = wmma_f16(a, b1, accB);
        __syncthreads();
    }

    const int d0 = dt * 16 + g * 8;
    v8h oa, ob;
    for (int r = 0; r < 8; ++r) { oa[r] = (half_t)accA[r]; ob[r] = (half_t)accB[r]; }
    *(v8h*)(ctx + ((size_t)(bh * 64 + etA * 16 + ln)) * 64 + d0) = oa;
    *(v8h*)(ctx + ((size_t)(bh * 64 + etB * 16 + ln)) * 64 + d0) = ob;
}

// ---------------------------------------------------------------------------
// attn[b][p][c] : out[p,e] = (qf[p,:] . ctx[:,e]) / max(qf[p,:].ksum, 1e-6)
// One block per (b, h, 128-p tile); qf+ctx staged in LDS, denom in f32.
// ---------------------------------------------------------------------------
__global__ __launch_bounds__(256)
void attn_kernel(const half_t* __restrict__ qf, const half_t* __restrict__ ctx,
                 const float* __restrict__ ksum, half_t* __restrict__ attn)
{
    __shared__ __align__(128) half_t sQ[128 * 72];
    __shared__ __align__(128) half_t sC[64 * 72];
    __shared__ float sS[64];
    __shared__ float sD[128];
    const int tid = threadIdx.x;
    const int lane = tid & 31, wave = tid >> 5;
    const int g = lane >> 4, ln = lane & 15;
    const int pB = blockIdx.x * 128, h = blockIdx.y, b = blockIdx.z;
    const int bh = b * HEADS + h;

    for (int i = 0; i < 4; ++i) {
        const int idx = tid + i * 256, row = idx >> 3, ch = idx & 7;
        *(v8h*)(sQ + row * 72 + ch * 8) =
            *(const v8h*)(qf + ((size_t)(bh * NPIX + pB + row)) * 64 + ch * 8);
    }
    for (int i = 0; i < 2; ++i) {
        const int idx = tid + i * 256, row = idx >> 3, ch = idx & 7;
        *(v8h*)(sC + row * 72 + ch * 8) =
            *(const v8h*)(ctx + ((size_t)(bh * 64 + row)) * 64 + ch * 8);
    }
    if (tid < 64) sS[tid] = ksum[(size_t)bh * 64 + tid];
    __syncthreads();

    if (tid < 128) {
        float s = 0.f;
        for (int dd = 0; dd < 8; ++dd) {
            v8h x = *(const v8h*)(sQ + tid * 72 + dd * 8);
            for (int j = 0; j < 8; ++j) s += (float)x[j] * sS[dd * 8 + j];
        }
        sD[tid] = fmaxf(s, 1e-6f);
    }
    __syncthreads();

    const v8f z8 = {0.f,0.f,0.f,0.f,0.f,0.f,0.f,0.f};
    v8f acc[4] = { z8, z8, z8, z8 };
#pragma unroll
    for (int ks = 0; ks < 2; ++ks) {
        const int p = wave * 16 + ln;
        const v8h alo = *(const v8h*)(sQ + p * 72 + ks * 32 + g * 8);
        const v8h ahi = *(const v8h*)(sQ + p * 72 + ks * 32 + 16 + g * 8);
        const v16h a = __builtin_shufflevector(alo, ahi,
                           0,1,2,3,4,5,6,7,8,9,10,11,12,13,14,15);
#pragma unroll
        for (int et = 0; et < 4; ++et) {
            const half_t* rp = sC + (et * 16 + ln) * 72 + ks * 32 + g * 8;
            const v8h blo = *(const v8h*)rp;
            const v8h bhi = *(const v8h*)(rp + 16);
            const v16h bb = __builtin_shufflevector(blo, bhi,
                                0,1,2,3,4,5,6,7,8,9,10,11,12,13,14,15);
            acc[et] = wmma_f16(a, bb, acc[et]);
        }
    }

    const int p0 = wave * 16 + g * 8;
    float den[8];
    {
        v4f d0 = *(const v4f*)(sD + p0);
        v4f d1 = *(const v4f*)(sD + p0 + 4);
        den[0]=d0.x; den[1]=d0.y; den[2]=d0.z; den[3]=d0.w;
        den[4]=d1.x; den[5]=d1.y; den[6]=d1.z; den[7]=d1.w;
    }
#pragma unroll
    for (int et = 0; et < 4; ++et) {
        const int c = h * 64 + et * 16 + ln;
        for (int r = 0; r < 8; ++r) {
            float o = acc[et][r] / den[r];
            o = (o != o) ? 0.f
                : (o ==  __builtin_inff() ?  10.f
                : (o == -__builtin_inff() ? -10.f : o));
            attn[((size_t)(b * NPIX + pB + p0 + r)) * CH + c] = (half_t)o;
        }
    }
}

// ---------------------------------------------------------------------------
// GroupNorm + residual + nan_to_num(+-20)
// ---------------------------------------------------------------------------
__global__ __launch_bounds__(256)
void gn_kernel(const float* __restrict__ proj, const float* __restrict__ stats,
               const float* __restrict__ resid, const float* __restrict__ gamma,
               const float* __restrict__ beta, float* __restrict__ outp)
{
    const size_t gid = (size_t)blockIdx.x * 256 + threadIdx.x;
    const size_t base = gid * 4;
    const int c = (int)((base >> 10) & 511);
    const int b = (int)(base >> 19);
    const int grp = c >> 4;
    const float sum = stats[(b * 32 + grp) * 2 + 0];
    const float ss  = stats[(b * 32 + grp) * 2 + 1];
    const float inv_n = 1.f / 16384.f;
    const float mu = sum * inv_n;
    float var = ss * inv_n - mu * mu;
    var = fmaxf(var, 0.f);
    const float rinv = rsqrtf(var + 1e-5f);
    const float ga = gamma[c], be = beta[c];
    v4f pv = *(const v4f*)(proj + base);
    v4f xv = *(const v4f*)(resid + base);
    v4f ov;
    for (int j = 0; j < 4; ++j) {
        float x = fminf(fmaxf(xv[j], -20.f), 20.f);
        float r = x + (pv[j] - mu) * rinv * ga + be;
        r = (r != r) ? 0.f
            : (r ==  __builtin_inff() ?  20.f
            : (r == -__builtin_inff() ? -20.f : r));
        ov[j] = r;
    }
    *(v4f*)(outp + base) = ov;
}

// ---------------------------------------------------------------------------
// Launch
// ---------------------------------------------------------------------------
// Workspace layout (bytes)
#define OFF_XR    0ull                     // 32 MB f16  [b][p][c] rain
#define OFF_XT    33554432ull              // 32 MB f16  [b][p][c] topo
#define OFF_W16   67108864ull              //  4 MB f16  8 x [512][512]
#define OFF_CTX   71303168ull              //  2 MB f16  [bh][e][d]
#define OFF_KSUM  73400320ull              // 64 KB f32
#define OFF_STATS 73465856ull              // 16 KB f32  (2 branches x [32][32][2])
#define OFF_KF    73482240ull              // 32 MB f16  kf / attn (aliased)
#define OFF_QF    107036672ull             // 32 MB f16  qf   } proj (f32, 64MB)
#define OFF_VV    140591104ull             // 32 MB f16  v    } aliases QF..VV

extern "C" void kernel_launch(void* const* d_in, const int* in_sizes, int n_in,
                              void* d_out, int out_size, void* d_ws, size_t ws_size,
                              hipStream_t stream)
{
    (void)in_sizes; (void)n_in; (void)out_size; (void)ws_size;
    const float* rain = (const float*)d_in[0];
    const float* topo = (const float*)d_in[1];
    char* ws = (char*)d_ws;

    half_t* xr16  = (half_t*)(ws + OFF_XR);
    half_t* xt16  = (half_t*)(ws + OFF_XT);
    half_t* w16   = (half_t*)(ws + OFF_W16);
    half_t* ctx16 = (half_t*)(ws + OFF_CTX);
    float*  ksum  = (float*)(ws + OFF_KSUM);
    float*  stats = (float*)(ws + OFF_STATS);
    half_t* kf    = (half_t*)(ws + OFF_KF);     // also attn (aliased, disjoint in time)
    half_t* qf    = (half_t*)(ws + OFF_QF);
    half_t* vv    = (half_t*)(ws + OFF_VV);
    float*  proj  = (float*)(ws + OFF_QF);      // 64 MB alias over qf+vv

    // Pre-passes: transpose+clip inputs to f16 [b][p][c]; weights f32->f16
    txc_kernel<<<dim3(32, 16, BATCH), 256, 0, stream>>>(rain, xr16);
    txc_kernel<<<dim3(32, 16, BATCH), 256, 0, stream>>>(topo, xt16);
    const int wIdx[8] = { 2, 4, 6, 8, 12, 14, 16, 18 };  // rq, tk, tv, rout, tq, rk, rv, tout
    for (int i = 0; i < 8; ++i)
        cvt_kernel<<<256, 256, 0, stream>>>((const float*)d_in[wIdx[i]],
                                            w16 + (size_t)i * 262144, 65536);
    zero_kernel<<<16, 256, 0, stream>>>(stats, 4096);

    for (int br = 0; br < 2; ++br) {
        const half_t* xq  = br ? xt16 : xr16;   // Q input
        const half_t* xkv = br ? xr16 : xt16;   // K/V input
        const half_t* wq  = w16 + (size_t)(br ? 4 : 0) * 262144;
        const half_t* wk  = w16 + (size_t)(br ? 5 : 1) * 262144;
        const half_t* wv  = w16 + (size_t)(br ? 6 : 2) * 262144;
        const half_t* wo  = w16 + (size_t)(br ? 7 : 3) * 262144;
        const float* bq = (const float*)d_in[br ? 13 :  3];
        const float* bk = (const float*)d_in[br ? 15 :  5];
        const float* bv = (const float*)d_in[br ? 17 :  7];
        const float* bo = (const float*)d_in[br ? 19 :  9];
        const float* gg = (const float*)d_in[br ? 20 : 10];
        const float* gb = (const float*)d_in[br ? 21 : 11];
        const float* resid = br ? topo : rain;
        float* outp = (float*)d_out + (size_t)br * (BATCH * CH * NPIX);
        float* st = stats + (size_t)br * 2048;

        gemm512_kernel<0><<<dim3(8, 4, BATCH), 256, 0, stream>>>(wq, xq,  bq, (void*)qf, nullptr);
        gemm512_kernel<1><<<dim3(8, 4, BATCH), 256, 0, stream>>>(wk, xkv, bk, (void*)kf, nullptr);
        gemm512_kernel<2><<<dim3(8, 4, BATCH), 256, 0, stream>>>(wv, xkv, bv, (void*)vv, nullptr);
        ksum_kernel<<<BATCH * HEADS, 256, 0, stream>>>(kf, ksum);
        ctx_kernel <<<BATCH * HEADS, 256, 0, stream>>>(kf, vv, ctx16);
        attn_kernel<<<dim3(8, HEADS, BATCH), 256, 0, stream>>>(qf, ctx16, ksum, kf /*attn alias*/);
        gemm512_kernel<3><<<dim3(8, 4, BATCH), 256, 0, stream>>>(wo, kf, bo, (void*)proj, st);
        gn_kernel<<<16384, 256, 0, stream>>>(proj, st, resid, gg, gb, outp);
    }
}
